// QNet_64888365908125
// MI455X (gfx1250) — matrix operand order repaired and checked by reading, not compiled, not run on hardware
//
#include <hip/hip_runtime.h>
#include <hip/hip_bf16.h>

#define NNODES 65536
#define DIM    128
#define ESG    262144
#define ECG    131072
#define NGRAPH 8
#define GSIZE  (NNODES / NGRAPH)

typedef __bf16 v16bf __attribute__((ext_vector_type(16)));
typedef float  v8f   __attribute__((ext_vector_type(8)));

union Frag { v16bf v; unsigned int u[8]; };

__device__ inline unsigned short f2bf(float f) {
    union { float f; unsigned u; } cv; cv.f = f;
    unsigned u = cv.u;
    u += 0x7FFFu + ((u >> 16) & 1u);          // round-to-nearest-even
    return (unsigned short)(u >> 16);
}

__device__ inline void atomAddF(float* p, float v) {
    unsafeAtomicAdd(p, v);                    // global_atomic_add_f32
}

// ---------------- elementwise / convert ----------------

__global__ __launch_bounds__(256) void conv_bf16_kernel(
    const float* __restrict__ s, unsigned short* __restrict__ d, int n) {
    int i = blockIdx.x * 256 + threadIdx.x;
    if (i < n) d[i] = f2bf(s[i]);
}

// h = relu(r0[row]*c0[col] + r1[row]*c1[col])   (rank-2 edge-embed init)
__global__ __launch_bounds__(256) void h0_rank2_kernel(
    const float* __restrict__ r0, const float* __restrict__ c0,
    const float* __restrict__ r1, const float* __restrict__ c1,
    float* __restrict__ out) {
    int i = blockIdx.x * 256 + threadIdx.x;
    int r = i >> 7, c = i & 127;
    float v = r0[r] * c0[c] + r1[r] * c1[c];
    out[i] = v > 0.f ? v : 0.f;
}

__global__ __launch_bounds__(256) void h0_rank1_kernel(
    const float* __restrict__ r0, const float* __restrict__ c0,
    float* __restrict__ out) {
    int i = blockIdx.x * 256 + threadIdx.x;
    int r = i >> 7, c = i & 127;
    float v = r0[r] * c0[c];
    out[i] = v > 0.f ? v : 0.f;
}

__global__ __launch_bounds__(256) void numc_kernel(
    const int* __restrict__ rowptr, float* __restrict__ out, int n) {
    int i = blockIdx.x * 256 + threadIdx.x;
    if (i < n) out[i] = (float)(rowptr[i + 1] - rowptr[i]);
}

// ---------------- CSR build (int atomics only) ----------------

__global__ __launch_bounds__(256) void sg_count_kernel(
    const int* __restrict__ dst, int* __restrict__ deg, int E) {
    int e = blockIdx.x * 256 + threadIdx.x;
    if (e < E) atomicAdd(&deg[dst[e]], 1);
}

__global__ __launch_bounds__(256) void cg_count_kernel(
    const int* __restrict__ eu, const int* __restrict__ ev, int* __restrict__ deg, int E) {
    int e = blockIdx.x * 256 + threadIdx.x;
    if (e < E) { atomicAdd(&deg[eu[e]], 1); atomicAdd(&deg[ev[e]], 1); }
}

__global__ __launch_bounds__(256) void ct_count_kernel(
    const int* __restrict__ cid, int* __restrict__ deg, int n) {
    int i = blockIdx.x * 256 + threadIdx.x;
    if (i < n) atomicAdd(&deg[cid[i]], 1);
}

// exclusive scan of degcur[0..n-1] -> rowptr (and degcur becomes the fill cursor)
__global__ __launch_bounds__(1024) void scan_kernel(
    int* __restrict__ degcur, int* __restrict__ rowptr, int n) {
    __shared__ int part[1024];
    int t = threadIdx.x;
    int cnt = n / 1024, base = t * cnt;
    int s = 0;
    for (int i = 0; i < cnt; ++i) s += degcur[base + i];
    part[t] = s;
    __syncthreads();
    for (int off = 1; off < 1024; off <<= 1) {
        int v = (t >= off) ? part[t - off] : 0;
        __syncthreads();
        part[t] += v;
        __syncthreads();
    }
    int run = part[t] - s;
    for (int i = 0; i < cnt; ++i) {
        int d = degcur[base + i];
        rowptr[base + i] = run;
        degcur[base + i] = run;
        run += d;
    }
    if (t == 1023) rowptr[n] = part[1023];
}

// fill sg CSR with src ids; accumulate per-node edge-feature sums (rank-2 x)
__global__ __launch_bounds__(256) void sg_fill_kernel(
    const int* __restrict__ src, const int* __restrict__ dst,
    const float* __restrict__ ef, int* __restrict__ cur, int* __restrict__ adj,
    float* __restrict__ wsum, float* __restrict__ xssum, int E) {
    int e = blockIdx.x * 256 + threadIdx.x;
    if (e >= E) return;
    int s = src[e], d = dst[e];
    int pos = atomicAdd(&cur[d], 1);
    adj[pos] = s;
    atomAddF(&wsum[d], ef[2 * e]);
    atomAddF(&xssum[d], ef[2 * e + 1]);
}

__global__ __launch_bounds__(256) void cg_fill_kernel(
    const int* __restrict__ eu, const int* __restrict__ ev,
    const float* __restrict__ cgw, int* __restrict__ cur, int* __restrict__ adj,
    float* __restrict__ wsum, int E) {
    int e = blockIdx.x * 256 + threadIdx.x;
    if (e >= E) return;
    int u = eu[e], v = ev[e];
    float w = cgw[e];
    int p0 = atomicAdd(&cur[v], 1); adj[p0] = u;
    int p1 = atomicAdd(&cur[u], 1); adj[p1] = v;
    atomAddF(&wsum[u], w);
    atomAddF(&wsum[v], w);
}

__global__ __launch_bounds__(256) void ct_fill_kernel(
    const int* __restrict__ cid, int* __restrict__ cur, int* __restrict__ adj, int n) {
    int i = blockIdx.x * 256 + threadIdx.x;
    if (i >= n) return;
    int pos = atomicAdd(&cur[cid[i]], 1);
    adj[pos] = i;
}

// ---------------- pull-gather (no float atomics; L2-bandwidth bound) ----------------
// wave per node, lane covers 4 cols: out[n,:] = sum_{s in adj[row(n)]} h[s,:]
__global__ __launch_bounds__(256) void csr_gather_kernel(
    const float* __restrict__ h, const int* __restrict__ rowptr,
    const int* __restrict__ adj, float* __restrict__ out) {
    int node = blockIdx.x * 8 + (threadIdx.x >> 5);
    int c = (threadIdx.x & 31) * 4;
    int beg = rowptr[node], end = rowptr[node + 1];
    float4 acc = make_float4(0.f, 0.f, 0.f, 0.f);
    for (int i = beg; i < end; ++i) {
        int s = adj[i];
        float4 v = *(const float4*)&h[(size_t)s * DIM + c];
        acc.x += v.x; acc.y += v.y; acc.z += v.z; acc.w += v.w;
    }
    *(float4*)&out[(size_t)node * DIM + c] = acc;
}

// ---------------- WMMA GEMM: Out = relu(A[M,K]@Wb[K,128] + epilogue) ----------------
// A from A0 (cols 0..127) and A1 (cols 128..255), f32 -> bf16 LDS (vectorized).
// EPI 0: + r0[row]*c0[col] + r1[row]*c1[col]   (sg: rank-2 x)
// EPI 1: + r0[row]*c0[col]                     (cg: rank-1 x)
// EPI 2: + r0[row]*c0[col] + bias[col]         (concat: num_contains column + bias)
template <int K, int EPI>
__global__ __launch_bounds__(256) void wmma_gemm_kernel(
    const float* __restrict__ A0, const float* __restrict__ A1,
    const unsigned short* __restrict__ Wb,
    const float* __restrict__ r0, const float* __restrict__ c0,
    const float* __restrict__ r1, const float* __restrict__ c1,
    const float* __restrict__ bias, float* __restrict__ Out) {
    __shared__ unsigned short sA[16 * K];
    constexpr int K4 = K / 4;
    const int row0 = blockIdx.x * 16;
    const int t = threadIdx.x;
    for (int e = t; e < 16 * K4; e += 256) {
        int r = e / K4, k4 = (e % K4) * 4;
        const float* sp = (K <= DIM || k4 < DIM)
                              ? &A0[(size_t)(row0 + r) * DIM + k4]
                              : &A1[(size_t)(row0 + r) * DIM + (k4 - DIM)];
        float4 f = *(const float4*)sp;
        unsigned lo = (unsigned)f2bf(f.x) | ((unsigned)f2bf(f.y) << 16);
        unsigned hi = (unsigned)f2bf(f.z) | ((unsigned)f2bf(f.w) << 16);
        *(uint2*)&sA[r * K + k4] = make_uint2(lo, hi);
    }
    __syncthreads();

    const int wave = t >> 5, lane = t & 31;
    const int hh = (lane >> 4) & 1, ml = lane & 15;
    v8f acc = {0.f, 0.f, 0.f, 0.f, 0.f, 0.f, 0.f, 0.f};
    const unsigned int* W32 = (const unsigned int*)Wb;

    for (int k0 = 0; k0 < K; k0 += 32) {
        Frag a, b;
#pragma unroll
        for (int n = 0; n < 8; ++n) {
            int kk = k0 + ((n < 4) ? 2 * n : 16 + 2 * (n - 4)) + (hh ? 8 : 0);
            a.u[n] = *(const unsigned int*)&sA[ml * K + kk];
        }
#pragma unroll
        for (int n = 0; n < 8; ++n) {
            int kb = k0 + ((n < 4) ? 2 * n : 16 + 2 * (n - 4)) + (hh ? 8 : 0);
            b.u[n] = (unsigned)Wb[kb * 128 + (wave << 4) + ml] | ((unsigned)Wb[(kb + 1) * 128 + (wave << 4) + ml] << 16);
        }
        acc = __builtin_amdgcn_wmma_f32_16x16x32_bf16(
            false, a.v, false, b.v, (short)0, acc, false, false);
    }

    const int col = (wave << 4) + ml;
#pragma unroll
    for (int r = 0; r < 8; ++r) {
        int m = r + (hh ? 8 : 0);
        int row = row0 + m;
        size_t idx = (size_t)row * DIM + col;
        float v = acc[r];
        if (EPI == 0)      v += r0[row] * c0[col] + r1[row] * c1[col];
        else if (EPI == 1) v += r0[row] * c0[col];
        else               v += r0[row] * c0[col] + bias[col];
        Out[idx] = v > 0.f ? v : 0.f;
    }
}

// ---------------- pooling / heads ----------------

__global__ __launch_bounds__(256) void pool_kernel(
    const float* __restrict__ ne, const unsigned char* __restrict__ valid,
    float* __restrict__ pool, float* __restrict__ ncg) {
    int g = blockIdx.x, t = threadIdx.x;
    int col = t & 127, rg = t >> 7;
    float s = 0.f, cnt = 0.f;
    int base = g * GSIZE;
    for (int r = rg; r < GSIZE; r += 2) {
        int node = base + r;
        if (valid[node]) {
            s += ne[(size_t)node * DIM + col];
            if (col == 0) cnt += 1.f;
        }
    }
    __shared__ float sp[256]; __shared__ float sc[2];
    sp[t] = s; if (col == 0) sc[rg] = cnt;
    __syncthreads();
    if (t < 128) {
        pool[g * DIM + t] = sp[t] + sp[t + 128];
        if (t == 0) ncg[g] = sc[0] + sc[1];
    }
}

__global__ __launch_bounds__(128) void heads_kernel(
    const float* __restrict__ pool, const float* __restrict__ ncg,
    const float* __restrict__ nodesW, const float* __restrict__ nodesB,
    const float* __restrict__ s1W, const float* __restrict__ s1b,
    const float* __restrict__ soutW, const float* __restrict__ soutb,
    float* __restrict__ ge, float* __restrict__ vg) {
    int j = threadIdx.x;
    __shared__ float red[128];
    for (int g = 0; g < NGRAPH; ++g) {
        float acc = nodesB[j];
        for (int k = 0; k < DIM; ++k) acc += pool[g * DIM + k] * nodesW[k * DIM + j];
        ge[g * DIM + j] = acc > 0.f ? acc : 0.f;
        float h = s1b[j];
        for (int k = 0; k < DIM; ++k) h += pool[g * DIM + k] * s1W[k * DIM + j];
        h += ncg[g] * s1W[DIM * DIM + j];
        h = h > 0.f ? h : 0.f;
        red[j] = h * soutW[j];
        __syncthreads();
        for (int off = 64; off; off >>= 1) {
            if (j < off) red[j] += red[j + off];
            __syncthreads();
        }
        if (j == 0) vg[g] = red[0] + soutb[0];
        __syncthreads();
    }
}

// act[e, 0:128]=ne[eu]+ne[ev]; [128:256]=graph_edge[g]; [256:384]=relu(w*weightW+wb)
__global__ __launch_bounds__(256) void act_assemble_kernel(
    const float* __restrict__ ne, const int* __restrict__ eu, const int* __restrict__ ev,
    const int* __restrict__ gid, const float* __restrict__ cgw,
    const float* __restrict__ ge, const float* __restrict__ wW,
    const float* __restrict__ wb, unsigned short* __restrict__ act, int E) {
    int tid = blockIdx.x * 256 + threadIdx.x;
    int e = tid >> 7, q = tid & 127;
    if (e >= E) return;
    int a = eu[e], b = ev[e], g = gid[e];
    float w = cgw[e];
    unsigned short* row = act + (size_t)e * 384;
    row[q]       = f2bf(ne[(size_t)a * DIM + q] + ne[(size_t)b * DIM + q]);
    row[128 + q] = f2bf(ge[g * DIM + q]);
    float we = w * wW[q] + wb[q];
    row[256 + q] = f2bf(we > 0.f ? we : 0.f);
}

// fused action head: hidden=relu(Act@a1W + w*a1W[384] + a1b); q = hidden.aoutW + aoutb + v[g]
__global__ __launch_bounds__(256) void action_gemm_kernel(
    const unsigned short* __restrict__ Act, const unsigned short* __restrict__ Wb,
    const float* __restrict__ cgw, const float* __restrict__ wlast,
    const float* __restrict__ bias, const float* __restrict__ aoutW,
    const float* __restrict__ aoutB, const float* __restrict__ vtab,
    const int* __restrict__ gid, float* __restrict__ Out) {
    __shared__ float sH[16 * 128];
    const int row0 = blockIdx.x * 16;
    const int t = threadIdx.x, wave = t >> 5, lane = t & 31;
    const int hh = (lane >> 4) & 1, ml = lane & 15;
    v8f acc = {0.f, 0.f, 0.f, 0.f, 0.f, 0.f, 0.f, 0.f};
    const unsigned int* A32 = (const unsigned int*)Act;
    const unsigned int* W32 = (const unsigned int*)Wb;

    for (int k0 = 0; k0 < 384; k0 += 32) {
        Frag a, b;
#pragma unroll
        for (int n = 0; n < 8; ++n) {
            int kk = k0 + ((n < 4) ? 2 * n : 16 + 2 * (n - 4)) + (hh ? 8 : 0);
            a.u[n] = A32[(size_t)(row0 + ml) * 192 + (kk >> 1)];
        }
#pragma unroll
        for (int n = 0; n < 8; ++n) {
            int kb = k0 + ((n < 4) ? 2 * n : 16 + 2 * (n - 4)) + (hh ? 8 : 0);
            b.u[n] = (unsigned)Wb[kb * 128 + (wave << 4) + ml] | ((unsigned)Wb[(kb + 1) * 128 + (wave << 4) + ml] << 16);
        }
        acc = __builtin_amdgcn_wmma_f32_16x16x32_bf16(
            false, a.v, false, b.v, (short)0, acc, false, false);
    }

    const int col = (wave << 4) + ml;
#pragma unroll
    for (int r = 0; r < 8; ++r) {
        int m = r + (hh ? 8 : 0);
        float v = acc[r] + cgw[row0 + m] * wlast[col] + bias[col];
        sH[m * DIM + col] = v > 0.f ? v : 0.f;
    }
    __syncthreads();
    if (t < 16) {
        float q = aoutB[0];
        for (int c = 0; c < DIM; ++c) q += sH[t * DIM + c] * aoutW[c];
        int row = row0 + t;
        Out[row] = q + vtab[gid[row]];
    }
}

// ---------------- host launch ----------------

extern "C" void kernel_launch(void* const* d_in, const int* in_sizes, int n_in,
                              void* d_out, int out_size, void* d_ws, size_t ws_size,
                              hipStream_t stream) {
    const float* sg_ef   = (const float*)d_in[0];
    const float* cg_w    = (const float*)d_in[1];
    const int*   sg_src  = (const int*)d_in[2];
    const int*   sg_dst  = (const int*)d_in[3];
    const int*   cg_eu   = (const int*)d_in[4];
    const int*   cg_ev   = (const int*)d_in[5];
    const int*   contain = (const int*)d_in[6];
    const unsigned char* valid = (const unsigned char*)d_in[8];
    const int*   eg      = (const int*)d_in[9];
    const float* We_sg   = (const float*)d_in[10];
    const float* Wc_sg   = (const float*)d_in[11];
    const float* We_cg   = (const float*)d_in[12];
    const float* Wc_cg   = (const float*)d_in[13];
    const float* concatW = (const float*)d_in[14];
    const float* concatB = (const float*)d_in[15];
    const float* weightW = (const float*)d_in[16];
    const float* weightB = (const float*)d_in[17];
    const float* nodesW  = (const float*)d_in[18];
    const float* nodesB  = (const float*)d_in[19];
    const float* s1W     = (const float*)d_in[20];
    const float* s1b     = (const float*)d_in[21];
    const float* soutW   = (const float*)d_in[22];
    const float* soutb   = (const float*)d_in[23];
    const float* a1W     = (const float*)d_in[24];
    const float* a1b     = (const float*)d_in[25];
    const float* aoutW   = (const float*)d_in[26];
    const float* aoutB   = (const float*)d_in[27];
    float* out = (float*)d_out;

    const size_t NB = (size_t)NNODES * DIM * 4;   // 32 MB per node-feature buffer
    char* ws = (char*)d_ws;
    float* h_sg   = (float*)(ws + 0 * NB);
    float* node_e = (float*)(ws + 1 * NB);
    float* agg    = (float*)(ws + 2 * NB);
    float* h_cg   = (float*)(ws + 3 * NB);
    float* mbs    = (float*)(ws + 4 * NB);
    unsigned short* act_bf = (unsigned short*)(ws + 2 * NB); // aliases agg+h_cg+mbs (dead)
    size_t off = 5 * NB;
    const size_t NI = (size_t)NNODES * 4;
    // zero-init region (single memset): 3 int cursors + 3 float sums
    char* zbase = ws + off;
    int*   cur_sg  = (int*)(ws + off);   off += NI;
    int*   cur_cg  = (int*)(ws + off);   off += NI;
    int*   cur_ct  = (int*)(ws + off);   off += NI;
    float* wsum_sg = (float*)(ws + off); off += NI;
    float* xssum   = (float*)(ws + off); off += NI;
    float* wsum_cg = (float*)(ws + off); off += NI;
    size_t zbytes = 6 * NI;
    int* rp_sg = (int*)(ws + off); off += NI + 64;
    int* rp_cg = (int*)(ws + off); off += NI + 64;
    int* rp_ct = (int*)(ws + off); off += NI + 64;
    int* adj_sg = (int*)(ws + off); off += (size_t)ESG * 4;
    int* adj_cg = (int*)(ws + off); off += (size_t)2 * ECG * 4;
    int* adj_ct = (int*)(ws + off); off += NI;
    float* numc = (float*)(ws + off); off += NI;
    float* pool = (float*)(ws + off); off += NGRAPH * DIM * 4;
    float* ncg  = (float*)(ws + off); off += NGRAPH * 4;
    float* ge   = (float*)(ws + off); off += NGRAPH * DIM * 4;
    float* vg   = (float*)(ws + off); off += NGRAPH * 4;
    unsigned short* Wc_sg_bf = (unsigned short*)(ws + off); off += DIM * DIM * 2;
    unsigned short* Wc_cg_bf = (unsigned short*)(ws + off); off += DIM * DIM * 2;
    unsigned short* cW_bf    = (unsigned short*)(ws + off); off += 256 * DIM * 2;
    unsigned short* a1_bf    = (unsigned short*)(ws + off); off += 384 * DIM * 2;

    // weights -> bf16
    conv_bf16_kernel<<<(DIM * DIM + 255) / 256, 256, 0, stream>>>(Wc_sg, Wc_sg_bf, DIM * DIM);
    conv_bf16_kernel<<<(DIM * DIM + 255) / 256, 256, 0, stream>>>(Wc_cg, Wc_cg_bf, DIM * DIM);
    conv_bf16_kernel<<<(256 * DIM + 255) / 256, 256, 0, stream>>>(concatW, cW_bf, 256 * DIM);
    conv_bf16_kernel<<<(384 * DIM + 255) / 256, 256, 0, stream>>>(a1W, a1_bf, 384 * DIM);

    // ---- CSR builds (int atomics only) ----
    hipMemsetAsync(zbase, 0, zbytes, stream);
    sg_count_kernel<<<ESG / 256, 256, 0, stream>>>(sg_dst, cur_sg, ESG);
    cg_count_kernel<<<ECG / 256, 256, 0, stream>>>(cg_eu, cg_ev, cur_cg, ECG);
    ct_count_kernel<<<NNODES / 256, 256, 0, stream>>>(contain, cur_ct, NNODES);
    scan_kernel<<<1, 1024, 0, stream>>>(cur_sg, rp_sg, NNODES);
    scan_kernel<<<1, 1024, 0, stream>>>(cur_cg, rp_cg, NNODES);
    scan_kernel<<<1, 1024, 0, stream>>>(cur_ct, rp_ct, NNODES);
    sg_fill_kernel<<<ESG / 256, 256, 0, stream>>>(sg_src, sg_dst, sg_ef, cur_sg, adj_sg,
                                                  wsum_sg, xssum, ESG);
    cg_fill_kernel<<<ECG / 256, 256, 0, stream>>>(cg_eu, cg_ev, cg_w, cur_cg, adj_cg,
                                                  wsum_cg, ECG);
    ct_fill_kernel<<<NNODES / 256, 256, 0, stream>>>(contain, cur_ct, adj_ct, NNODES);
    numc_kernel<<<NNODES / 256, 256, 0, stream>>>(rp_ct, numc, NNODES);

    // ---- state-graph GNN: h = relu(x);  x = rank-2(wsum,xssum) ----
    h0_rank2_kernel<<<(NNODES * DIM) / 256, 256, 0, stream>>>(
        wsum_sg, We_sg, xssum, We_sg + DIM, h_sg);
    for (int lv = 0; lv < 3; ++lv) {
        csr_gather_kernel<<<NNODES / 8, 256, 0, stream>>>(h_sg, rp_sg, adj_sg, agg);
        wmma_gemm_kernel<128, 0><<<NNODES / 16, 256, 0, stream>>>(
            agg, nullptr, Wc_sg_bf, wsum_sg, We_sg, xssum, We_sg + DIM, nullptr, h_sg);
    }

    // ---- MBS pull-gather (contain-group CSR) ----
    csr_gather_kernel<<<NNODES / 8, 256, 0, stream>>>(h_sg, rp_ct, adj_ct, mbs);

    // ---- contracted-graph GNN (bidirected) ----
    h0_rank1_kernel<<<(NNODES * DIM) / 256, 256, 0, stream>>>(wsum_cg, We_cg, h_cg);
    for (int lv = 0; lv < 2; ++lv) {
        csr_gather_kernel<<<NNODES / 8, 256, 0, stream>>>(h_cg, rp_cg, adj_cg, agg);
        wmma_gemm_kernel<128, 1><<<NNODES / 16, 256, 0, stream>>>(
            agg, nullptr, Wc_cg_bf, wsum_cg, We_cg, nullptr, nullptr, nullptr, h_cg);
    }

    // ---- node_embed = relu([h_cg, mbs, numc] @ concat_W + b) ----
    wmma_gemm_kernel<256, 2><<<NNODES / 16, 256, 0, stream>>>(
        h_cg, mbs, cW_bf, numc, concatW + 256 * DIM, nullptr, nullptr, concatB, node_e);

    // ---- pooling + small heads ----
    pool_kernel<<<NGRAPH, 256, 0, stream>>>(node_e, valid, pool, ncg);
    heads_kernel<<<1, 128, 0, stream>>>(pool, ncg, nodesW, nodesB, s1W, s1b, soutW, soutb, ge, vg);

    // ---- action head (act_bf aliases the now-dead agg/h_cg/mbs region) ----
    act_assemble_kernel<<<(ECG * 128) / 256, 256, 0, stream>>>(
        node_e, cg_eu, cg_ev, eg, cg_w, ge, weightW, weightB, act_bf, ECG);
    action_gemm_kernel<<<ECG / 16, 256, 0, stream>>>(
        act_bf, a1_bf, cg_w, a1W + 384 * DIM, a1b, aoutW, aoutB, vg, eg, out);

    (void)in_sizes; (void)n_in; (void)out_size; (void)ws_size;
}
